// CSA_74818330296427
// MI455X (gfx1250) — matrix operand order repaired and checked by reading, not compiled, not run on hardware
//
#include <hip/hip_runtime.h>
#include <hip/hip_bf16.h>
#include <math.h>

// ---------------------------------------------------------------------------
// MI455X (gfx1250) multi-head attention, bf16 WMMA path.
// B=4, SQ=1024, SKV=2048, DIM_Q=DIM_KV=512, H=8, DH=64, DIM_OUT=256.
// ---------------------------------------------------------------------------

typedef __attribute__((ext_vector_type(16))) __bf16 v16bf;
typedef __attribute__((ext_vector_type(8)))  float  v8f;
typedef __attribute__((ext_vector_type(4)))  float  f4;

#define NEGV (-1e9f)

__device__ __forceinline__ v16bf cvt16(const float* f) {
    v16bf r;
#pragma unroll
    for (int i = 0; i < 16; ++i) r[i] = (__bf16)f[i];
    return r;
}

// Load 8 contiguous floats (two b128 loads).
__device__ __forceinline__ void ld8(const float* p, float* dst) {
    float4 x = *(const float4*)p;
    float4 y = *(const float4*)(p + 4);
    dst[0] = x.x; dst[1] = x.y; dst[2] = x.z; dst[3] = x.w;
    dst[4] = y.x; dst[5] = y.y; dst[6] = y.z; dst[7] = y.w;
}

__device__ __forceinline__ v8f wmma_bf16(v16bf a, v16bf b, v8f c) {
    return __builtin_amdgcn_wmma_f32_16x16x32_bf16(
        /*neg_a=*/false, a, /*neg_b=*/false, b,
        /*c_mod=*/(short)0, c, /*reuse_a=*/false, /*reuse_b=*/false);
}

// ---------------------------------------------------------------------------
// Projection GEMM: C[M,N] = A[M,K] @ W[N,K]^T + bias[N]
// One wave computes a 64x32 block (4 M-tiles x 2 N-tiles of 16x16):
// 8 WMMAs per K-chunk from 6 fragment loads -> ~2.7x fewer loads/cvts per
// WMMA than unblocked, while all operands stay < 150 VGPRs.
// ---------------------------------------------------------------------------
__global__ __launch_bounds__(256)
void proj_gemm(const float* __restrict__ A, const float* __restrict__ W,
               const float* __restrict__ bias, float* __restrict__ C,
               int M, int N, int K) {
    int wid  = blockIdx.x * (blockDim.x >> 5) + (threadIdx.x >> 5);
    int lane = threadIdx.x & 31;
    int tn = N >> 5;                           // 32-wide column blocks
    int tile_m = wid / tn;                     // 64-row block
    int tile_n = wid % tn;
    if (tile_m >= (M >> 6)) return;            // wave-uniform guard

    int m16   = lane & 15;
    int baseA = (lane < 16) ? 0 : 16;          // A-frag K offset (contiguous k = 16h + s)
    int baseB = (lane < 16) ? 0 : 16;          // B-frag K offset (contiguous k = 16h + s, same as A)

    const float* arow[4];
    const float* wrow[2];
#pragma unroll
    for (int i = 0; i < 4; ++i)
        arow[i] = A + (size_t)(tile_m * 64 + i * 16 + m16) * K;
#pragma unroll
    for (int j = 0; j < 2; ++j)
        wrow[j] = W + (size_t)(tile_n * 32 + j * 16 + m16) * K;

    v8f acc[4][2];
#pragma unroll
    for (int i = 0; i < 4; ++i)
#pragma unroll
        for (int j = 0; j < 2; ++j)
            acc[i][j] = (v8f){};

    for (int k0 = 0; k0 < K; k0 += 32) {
        v16bf af[4], bf[2];
#pragma unroll
        for (int i = 0; i < 4; ++i) {
            float fa[16];
            ld8(arow[i] + k0 + baseA,      fa);
            ld8(arow[i] + k0 + baseA + 8, fa + 8);
            af[i] = cvt16(fa);
        }
#pragma unroll
        for (int j = 0; j < 2; ++j) {
            float fb[16];
            ld8(wrow[j] + k0 + baseB,     fb);
            ld8(wrow[j] + k0 + baseB + 8, fb + 8);
            bf[j] = cvt16(fb);
        }
#pragma unroll
        for (int i = 0; i < 4; ++i)
#pragma unroll
            for (int j = 0; j < 2; ++j)
                acc[i][j] = wmma_bf16(af[i], bf[j], acc[i][j]);
    }

    int rbase = (lane < 16) ? 0 : 8;
#pragma unroll
    for (int j = 0; j < 2; ++j) {
        int col   = tile_n * 32 + j * 16 + m16;
        float bvv = bias[col];
#pragma unroll
        for (int i = 0; i < 4; ++i) {
#pragma unroll
            for (int r = 0; r < 8; ++r) {
                int row = tile_m * 64 + i * 16 + rbase + r;
                C[(size_t)row * N + col] = acc[i][j][r] + bvv;
            }
        }
    }
}

// ---------------------------------------------------------------------------
// Attention core: one block per (b, h, 16-query tile).
//  Pass 1: scores (bf16 WMMA) + masks -> LDS [16][2048]
//  Pass 2: softmax + normalized weights write (nontemporal float4 stream-out)
//  Pass 3: P @ V (bf16 WMMA from LDS) -> att_concat [B,SQ,512]
// ---------------------------------------------------------------------------
__global__ __launch_bounds__(256)
void attn_kernel(const float* __restrict__ Qp, const float* __restrict__ Kp,
                 const float* __restrict__ Vp, const float* __restrict__ aff,
                 const int* __restrict__ qbin, const int* __restrict__ kvmask,
                 float* __restrict__ wout, float* __restrict__ attc) {
    __shared__ float sS[16][2048];   // 128 KB score strip (of 320 KB WGP LDS)
    __shared__ float red[256];
    __shared__ float rowmax[16];
    __shared__ float rowsum[16];

    const int tid  = threadIdx.x;
    const int lane = tid & 31;
    const int wave = tid >> 5;
    const int qt = blockIdx.x & 63;          // 1024/16 query tiles
    const int h  = (blockIdx.x >> 6) & 7;
    const int b  = blockIdx.x >> 9;

    const int m16   = lane & 15;
    const int hi    = (lane < 16) ? 0 : 1;
    const int baseA = hi ? 16 : 0;
    const int baseB = hi ? 16 : 0;

    // --- Q fragments for this tile (Dh = 64 -> two 32-wide K chunks), loaded once
    const float* qrow = Qp + ((size_t)(b * 1024 + qt * 16 + m16)) * 512 + h * 64;
    float fq[32];
    ld8(qrow + baseA,           fq);
    ld8(qrow + baseA + 8,       fq + 8);
    ld8(qrow + 32 + baseA,      fq + 16);
    ld8(qrow + 32 + baseA + 8,  fq + 24);
    v16bf aQ0 = cvt16(fq);
    v16bf aQ1 = cvt16(fq + 16);

    // --- Pass 1: scores + masks, strip into LDS
    for (int t = wave; t < 128; t += 8) {
        const int kvg = t * 16 + m16;        // this lane's kv column
        const float* krow = Kp + ((size_t)(b * 2048 + kvg)) * 512 + h * 64;
        if (t + 8 < 128) {                   // gfx1250 global_prefetch path
            const float* nrow = Kp + ((size_t)(b * 2048 + kvg + 128)) * 512 + h * 64;
            __builtin_prefetch(nrow, 0, 1);
        }
        float fb[32];
        ld8(krow + baseB,          fb);
        ld8(krow + baseB + 8,      fb + 8);
        ld8(krow + 32 + baseB,     fb + 16);
        ld8(krow + 32 + baseB + 8, fb + 24);

        v8f acc = {};
        acc = wmma_bf16(aQ0, cvt16(fb),      acc);
        acc = wmma_bf16(aQ1, cvt16(fb + 16), acc);

        const int kvm = kvmask[b * 2048 + kvg];
#pragma unroll
        for (int r = 0; r < 8; ++r) {
            int ql = r + (hi ? 8 : 0);
            int qg = qt * 16 + ql;
            float s = acc[r] * 0.125f;                 // 1/sqrt(64)
            if (kvm) s = NEGV;                          // kv masked_fill
            size_t idx = ((size_t)(b * 1024 + qg)) * 2048 + kvg;
            // masks are single-touch streams: keep them out of L2 (TH=NT)
            s *= __builtin_nontemporal_load(&aff[idx]); // multiplicative affinity
            if (__builtin_nontemporal_load(&qbin[idx])) s = NEGV; // binary mask
            sS[ql][kvg] = s;
        }
    }
    __syncthreads();

    // --- Pass 2: softmax over each of 16 rows (16 threads / row)
    const int row = tid >> 4;
    const int l16 = tid & 15;
    float mx = -INFINITY;
    for (int i = l16; i < 2048; i += 16) mx = fmaxf(mx, sS[row][i]);
    red[tid] = mx;
    __syncthreads();
    if (l16 == 0) {
        float mm = red[tid];
#pragma unroll
        for (int j = 1; j < 16; ++j) mm = fmaxf(mm, red[tid + j]);
        rowmax[row] = mm;
    }
    __syncthreads();
    const float mrow = rowmax[row];
    float sum = 0.f;
    for (int i = l16; i < 2048; i += 16) {
        float e = __expf(sS[row][i] - mrow);
        sS[row][i] = e;
        sum += e;
    }
    red[tid] = sum;
    __syncthreads();
    if (l16 == 0) {
        float ss = 0.f;
#pragma unroll
        for (int j = 0; j < 16; ++j) ss += red[tid + j];
        rowsum[row] = ss;
    }
    __syncthreads();

    // normalize in LDS + single coalesced nontemporal stream-out of the weights
    float* wbase = wout + (((size_t)(b * 8 + h)) * 1024 + qt * 16) * 2048;
    for (int idx = tid; idx < 8192; idx += 256) {
        int flat = idx << 2;
        int r = flat >> 11;
        int c = flat & 2047;
        float inv = 1.0f / rowsum[r];
        f4 v = *(const f4*)&sS[r][c];
        v *= inv;
        *(f4*)&sS[r][c] = v;
        __builtin_nontemporal_store(v, (f4*)&wbase[(size_t)r * 2048 + c]);
    }
    __syncthreads();

    // --- Pass 3: att = P(16x2048) @ Vh(2048x64); waves 0..3 each own 16 columns
    if (wave < 4) {
        v8f acc = {};
        const int d = h * 64 + wave * 16 + m16;       // this lane's output column
        for (int c = 0; c < 64; ++c) {
            const int k0 = c * 32;
            float fa[16], fb[16];
            const float* rowp = &sS[m16][k0];
            ld8(rowp + baseA,      fa);
            ld8(rowp + baseA + 8,  fa + 8);
            const int kvb = b * 2048 + k0 + baseB;
#pragma unroll
            for (int e = 0; e < 16; ++e)
                fb[e] = Vp[((size_t)(kvb + e)) * 512 + d];
            acc = wmma_bf16(cvt16(fa), cvt16(fb), acc);
        }
#pragma unroll
        for (int r = 0; r < 8; ++r) {
            int ql = r + (hi ? 8 : 0);
            attc[((size_t)(b * 1024 + qt * 16 + ql)) * 512 + h * 64 + wave * 16 + m16] = acc[r];
        }
    }
}

// ---------------------------------------------------------------------------
extern "C" void kernel_launch(void* const* d_in, const int* in_sizes, int n_in,
                              void* d_out, int out_size, void* d_ws, size_t ws_size,
                              hipStream_t stream) {
    const float* Q   = (const float*)d_in[0];
    const float* K   = (const float*)d_in[1];
    const float* V   = (const float*)d_in[2];
    const float* aff = (const float*)d_in[3];
    const int*   qbn = (const int*)d_in[4];
    const int*   kvm = (const int*)d_in[5];
    const float* Wq  = (const float*)d_in[6];
    const float* bq  = (const float*)d_in[7];
    const float* Wk  = (const float*)d_in[8];
    const float* bk  = (const float*)d_in[9];
    const float* Wv  = (const float*)d_in[10];
    const float* bv  = (const float*)d_in[11];
    const float* Wo  = (const float*)d_in[12];
    const float* bo  = (const float*)d_in[13];

    float* out     = (float*)d_out;
    float* att_out = out;                               // [4,1024,256]
    float* w_out   = out + (size_t)4 * 1024 * 256;      // [4,8,1024,2048]

    float* ws = (float*)d_ws;
    float* Qp = ws;                                     // 4096*512
    float* Kp = Qp + (size_t)4096 * 512;                // 8192*512
    float* Vp = Kp + (size_t)8192 * 512;                // 8192*512
    float* Ac = Vp + (size_t)8192 * 512;                // 4096*512 (concat heads)

    dim3 blk(256);
    // Q/K/V projections: one wave per 64x32 block, 8 waves per block.
    proj_gemm<<<(4096 / 64) * (512 / 32) / 8, blk, 0, stream>>>(Q, Wq, bq, Qp, 4096, 512, 512);
    proj_gemm<<<(8192 / 64) * (512 / 32) / 8, blk, 0, stream>>>(K, Wk, bk, Kp, 8192, 512, 512);
    proj_gemm<<<(8192 / 64) * (512 / 32) / 8, blk, 0, stream>>>(V, Wv, bv, Vp, 8192, 512, 512);
    // Attention: one block per (b, h, q-tile) = 4*8*64 blocks.
    attn_kernel<<<4 * 8 * 64, blk, 0, stream>>>(Qp, Kp, Vp, aff, qbn, kvm, w_out, Ac);
    // Output projection -> first segment of d_out.
    proj_gemm<<<(4096 / 64) * (256 / 32) / 8, blk, 0, stream>>>(Ac, Wo, bo, att_out, 4096, 256, 512);
}
